// GPT_model_40398462386806
// MI455X (gfx1250) — matrix operand order repaired and checked by reading, not compiled, not run on hardware
//
#include <hip/hip_runtime.h>
#include <hip/hip_bf16.h>

// ---------------------------------------------------------------------------
// GPT forward + loss for MI455X (gfx1250).
// All GEMMs: v_wmma_f32_16x16x32_bf16, bf16 operands pre-converted/pre-transposed,
// LDS tiles double-buffered and filled with global_load_async_to_lds_b128
// (ASYNCcnt) + s_wait_asynccnt.
// ---------------------------------------------------------------------------

#ifndef USE_ASYNC_LDS
#define USE_ASYNC_LDS 1
#endif

typedef __attribute__((ext_vector_type(16))) __bf16 v16bf;
typedef __attribute__((ext_vector_type(8)))  __bf16 v8bf;
typedef __attribute__((ext_vector_type(8)))  float  v8f;

__device__ __forceinline__ __bf16 f2bf(float f) {
    union { float f; unsigned u; } in; in.f = f;
    unsigned r = in.u + 0x7FFFu + ((in.u >> 16) & 1u);   // round-to-nearest-even
    union { unsigned short s; __bf16 b; } out;
    out.s = (unsigned short)(r >> 16);
    return out.b;
}

// 16-byte global -> LDS copy. Async path uses the CDNA5 ASYNC pipe: data never
// touches VGPRs; completion tracked by ASYNCcnt. LDS byte offset = low 32 bits
// of the generic pointer (LDS aperture is in the high word).
__device__ __forceinline__ void g2lds_b128(void* lds, const void* g) {
#if USE_ASYNC_LDS
    unsigned off = (unsigned)(unsigned long long)lds;
    asm volatile("global_load_async_to_lds_b128 %0, %1, off"
                 :: "v"(off), "v"(g) : "memory");
#else
    *(uint4*)lds = *(const uint4*)g;
#endif
}
__device__ __forceinline__ void g2lds_wait() {
#if USE_ASYNC_LDS
    asm volatile("s_wait_asynccnt 0x0" ::: "memory");
#endif
}

#define BM 128
#define BN 64
#define BK 32
#define LDP (BK + 8)   // LDS row stride: 40 bf16 = 80B, keeps 16B alignment

// C[M,N] = scale * (A[M,K] @ B^T) + bias  (B given as [N,K] row-major).
// Optional fp32 and/or bf16 outputs, ReLU, causal+pad score mask.
__global__ __launch_bounds__(256)
void gemm_bf16_wmma(const __bf16* __restrict__ A, long lda, long aOff,
                    const __bf16* __restrict__ Bm, long ldb, long bOff,
                    float* __restrict__ Cf, __bf16* __restrict__ Cb,
                    long ldc, long cOff, int K,
                    const float* __restrict__ bias,
                    float scale, int relu,
                    int maskMode, const int* __restrict__ tok)
{
    __shared__ __align__(16) __bf16 As[2][BM][LDP];
    __shared__ __align__(16) __bf16 Bs[2][BN][LDP];

    const int t    = threadIdx.x;
    const int lane = t & 31;
    const int wave = t >> 5;
    const int wm   = wave & 3;        // 4 wave-rows * 32
    const int wn   = wave >> 2;       // 2 wave-cols * 32
    const int row0 = blockIdx.x * BM;
    const int col0 = blockIdx.y * BN;

    const __bf16* Ab = A  + aOff;
    const __bf16* Bb = Bm + bOff;

    v8f acc[2][2] = {};

    const int lm  = lane & 15;
    const int kbA = (lane < 16) ? 0 : 16;   // A frag: K contiguous per half-wave
    const int kbB = (lane < 16) ? 0 : 16;   // B frag: K contiguous per half-wave

    // one staging round = pure 16B copies: A 512 chunks (2/thread), B 256 (1/thread)
    auto stage = [&](int buf, int kt) {
        #pragma unroll
        for (int i = 0; i < 2; ++i) {
            int f = i * 256 + t;
            int r = f >> 2, c8 = (f & 3) * 8;
            g2lds_b128(&As[buf][r][c8], &Ab[(long)(row0 + r) * lda + kt + c8]);
        }
        {
            int n = t >> 2, c8 = (t & 3) * 8;
            g2lds_b128(&Bs[buf][n][c8], &Bb[(long)(col0 + n) * ldb + kt + c8]);
        }
    };

    stage(0, 0);
    int cur = 0;
    for (int kt = 0; kt < K; kt += BK) {
        g2lds_wait();            // my async copies for As/Bs[cur] done
        __syncthreads();         // everyone's copies visible
        if (kt + BK < K) stage(cur ^ 1, kt + BK);   // overlap next tile

        v16bf af[2], bfrag[2];
        #pragma unroll
        for (int mi = 0; mi < 2; ++mi) {
            int r = wm * 32 + mi * 16 + lm;
            v8bf lo = *(const v8bf*)&As[cur][r][kbA];
            v8bf hi = *(const v8bf*)&As[cur][r][kbA + 8];
            af[mi] = __builtin_shufflevector(lo, hi, 0,1,2,3,4,5,6,7,8,9,10,11,12,13,14,15);
        }
        #pragma unroll
        for (int ni = 0; ni < 2; ++ni) {
            int n = wn * 32 + ni * 16 + lm;
            v8bf lo = *(const v8bf*)&Bs[cur][n][kbB];
            v8bf hi = *(const v8bf*)&Bs[cur][n][kbB + 8];
            bfrag[ni] = __builtin_shufflevector(lo, hi, 0,1,2,3,4,5,6,7,8,9,10,11,12,13,14,15);
        }
        #pragma unroll
        for (int mi = 0; mi < 2; ++mi)
            #pragma unroll
            for (int ni = 0; ni < 2; ++ni)
                acc[mi][ni] = __builtin_amdgcn_wmma_f32_16x16x32_bf16(
                    false, af[mi], false, bfrag[ni], (short)0, acc[mi][ni], false, false);
        cur ^= 1;
    }

    // epilogue: scale, bias, relu, score mask; fp32 and/or bf16 stores
    const int rsub = (lane < 16) ? 0 : 8;
    #pragma unroll
    for (int mi = 0; mi < 2; ++mi)
        #pragma unroll
        for (int ni = 0; ni < 2; ++ni) {
            int cg = col0 + wn * 32 + ni * 16 + lm;
            float bv = bias ? bias[cg] : 0.f;
            #pragma unroll
            for (int r = 0; r < 8; ++r) {
                int rg = row0 + wm * 32 + mi * 16 + rsub + r;
                float val = acc[mi][ni][r] * scale + bv;
                if (relu) val = fmaxf(val, 0.f);
                if (maskMode && (cg > rg || tok[rg] == 0)) val = -1e9f;
                long idx = cOff + (long)rg * ldc + cg;
                if (Cf) Cf[idx] = val;
                if (Cb) Cb[idx] = f2bf(val);
            }
        }
}

// dst[c*lddst + r] = bf16(src[r*ldsrc + c])  — tiled transpose+convert (weights)
__global__ __launch_bounds__(256)
void transpose_f32_bf16(const float* __restrict__ src, long ldsrc,
                        __bf16* __restrict__ dst, long lddst)
{
    __shared__ float tile[32][33];
    int r0 = blockIdx.y * 32, c0 = blockIdx.x * 32;
    int tx = threadIdx.x & 31, ty = threadIdx.x >> 5;
    #pragma unroll
    for (int i = 0; i < 4; ++i)
        tile[ty + i * 8][tx] = src[(long)(r0 + ty + i * 8) * ldsrc + c0 + tx];
    __syncthreads();
    #pragma unroll
    for (int i = 0; i < 4; ++i)
        dst[(long)(c0 + ty + i * 8) * lddst + r0 + tx] = f2bf(tile[tx][ty + i * 8]);
}

// vT[bh][d][s] = vb[b][s][h*DH+d]   (bf16 tiled transpose, batched over b*H)
__global__ __launch_bounds__(256)
void transpose_v_bf16(const __bf16* __restrict__ vb, __bf16* __restrict__ vT,
                      int S, int D, int DH, int H)
{
    __shared__ __bf16 tile[32][33];
    int z = blockIdx.z;
    const __bf16* src = vb + (long)(z / H) * S * D + (long)(z % H) * DH;
    __bf16* dst = vT + (long)z * DH * S;
    int r0 = blockIdx.y * 32, c0 = blockIdx.x * 32;    // r = s, c = d
    int tx = threadIdx.x & 31, ty = threadIdx.x >> 5;
    #pragma unroll
    for (int i = 0; i < 4; ++i)
        tile[ty + i * 8][tx] = src[(long)(r0 + ty + i * 8) * D + c0 + tx];
    __syncthreads();
    #pragma unroll
    for (int i = 0; i < 4; ++i)
        dst[(long)(c0 + ty + i * 8) * S + r0 + tx] = tile[tx][ty + i * 8];
}

// h = tok_emb[x] + pos_emb  (fp32 + bf16 copies)
__global__ void embed_kernel(const int* __restrict__ x, const float* __restrict__ tok_emb,
                             const float* __restrict__ pos_emb,
                             float* __restrict__ h, __bf16* __restrict__ hb,
                             int S, int D, long total)
{
    long i = (long)blockIdx.x * blockDim.x + threadIdx.x;
    if (i >= total) return;
    int d = (int)(i % D);
    long bs = i / D;
    int s = (int)(bs % S);
    float v = tok_emb[(long)x[bs] * D + d] + pos_emb[(long)s * D + d];
    h[i] = v; hb[i] = f2bf(v);
}

// softmax over the QUERY axis (reference axis=2); emits bf16 for the A·V GEMM
__global__ __launch_bounds__(256)
void softmax_q_kernel(const float* __restrict__ P, __bf16* __restrict__ Pb, int S)
{
    int col = blockIdx.x * blockDim.x + threadIdx.x;
    if (col >= S) return;
    float m = -1e30f;
    for (int q = 0; q < S; ++q) m = fmaxf(m, P[(long)q * S + col]);
    float s = 0.f;
    for (int q = 0; q < S; ++q) s += __expf(P[(long)q * S + col] - m);
    float inv = 1.f / s;
    for (int q = 0; q < S; ++q) {
        long idx = (long)q * S + col;
        Pb[idx] = f2bf(__expf(P[idx] - m) * inv);
    }
}

// out = LayerNorm(x + res) * g + b ; writes fp32 out and bf16 copy
__global__ __launch_bounds__(256)
void add_ln_kernel(const float* __restrict__ x, const float* __restrict__ res,
                   const float* __restrict__ g, const float* __restrict__ b,
                   float* __restrict__ out, __bf16* __restrict__ outb, int D)
{
    long row = blockIdx.x;
    __shared__ float s1[256], s2[256];
    float v[3];
    float sum = 0.f, sq = 0.f;
    #pragma unroll
    for (int i = 0; i < 3; ++i) {
        int d = threadIdx.x + i * 256;
        float tv = x[row * D + d] + res[row * D + d];
        v[i] = tv; sum += tv; sq += tv * tv;
    }
    s1[threadIdx.x] = sum; s2[threadIdx.x] = sq;
    __syncthreads();
    for (int o = 128; o > 0; o >>= 1) {
        if (threadIdx.x < o) { s1[threadIdx.x] += s1[threadIdx.x + o];
                               s2[threadIdx.x] += s2[threadIdx.x + o]; }
        __syncthreads();
    }
    float mean = s1[0] / D;
    float var  = s2[0] / D - mean * mean;
    float inv  = rsqrtf(var + 1e-5f);
    #pragma unroll
    for (int i = 0; i < 3; ++i) {
        int d = threadIdx.x + i * 256;
        float o = (v[i] - mean) * inv * g[d] + b[d];
        out[row * D + d] = o;
        outb[row * D + d] = f2bf(o);
    }
}

__global__ void lse_init_kernel(float* rowM, float* rowS, float* rowL, int R)
{
    int i = blockIdx.x * blockDim.x + threadIdx.x;
    if (i < R) { rowM[i] = -1e30f; rowS[i] = 0.f; rowL[i] = 0.f; }
}

// streaming log-sum-exp over classifier chunks + grab logits[label]
__global__ __launch_bounds__(256)
void lse_chunk_kernel(const float* __restrict__ chunk, int CN, int c0,
                      const int* __restrict__ label,
                      float* __restrict__ rowM, float* __restrict__ rowS,
                      float* __restrict__ rowL)
{
    long row = blockIdx.x;
    const float* p = chunk + row * CN;
    __shared__ float sh[256];
    float m = -1e30f;
    for (int c = threadIdx.x; c < CN; c += 256) m = fmaxf(m, p[c]);
    sh[threadIdx.x] = m; __syncthreads();
    for (int o = 128; o > 0; o >>= 1) {
        if (threadIdx.x < o) sh[threadIdx.x] = fmaxf(sh[threadIdx.x], sh[threadIdx.x + o]);
        __syncthreads();
    }
    m = sh[0]; __syncthreads();
    float s = 0.f;
    for (int c = threadIdx.x; c < CN; c += 256) s += __expf(p[c] - m);
    sh[threadIdx.x] = s; __syncthreads();
    for (int o = 128; o > 0; o >>= 1) {
        if (threadIdx.x < o) sh[threadIdx.x] += sh[threadIdx.x + o];
        __syncthreads();
    }
    if (threadIdx.x == 0) {
        float sc = sh[0];
        float Mo = rowM[row];
        float Mn = fmaxf(Mo, m);
        rowS[row] = rowS[row] * __expf(Mo - Mn) + sc * __expf(m - Mn);
        rowM[row] = Mn;
    }
    if (threadIdx.x == 1) {
        int l = label[row];
        if (l >= c0 && l < c0 + CN) rowL[row] = p[l - c0];
    }
}

__global__ __launch_bounds__(256)
void loss_kernel(const float* __restrict__ rowM, const float* __restrict__ rowS,
                 const float* __restrict__ rowL, float* __restrict__ out, int R)
{
    __shared__ float sh[256];
    float s = 0.f;
    for (int r = threadIdx.x; r < R; r += 256)
        s += rowM[r] + __logf(rowS[r]) - rowL[r];
    sh[threadIdx.x] = s; __syncthreads();
    for (int o = 128; o > 0; o >>= 1) {
        if (threadIdx.x < o) sh[threadIdx.x] += sh[threadIdx.x + o];
        __syncthreads();
    }
    if (threadIdx.x == 0) out[0] = sh[0] / (float)R;
}

// ---------------------------------------------------------------------------

extern "C" void kernel_launch(void* const* d_in, const int* in_sizes, int n_in,
                              void* d_out, int out_size, void* d_ws, size_t ws_size,
                              hipStream_t stream)
{
    (void)in_sizes; (void)n_in; (void)out_size;
    const int B = 4, S = 2048, D = 768, H = 4, V = 32000, Nl = 3, FF = 1024, DH = 192;
    const long M = (long)B * S;                 // 8192

    const int*   x       = (const int*)d_in[0];
    const int*   label   = (const int*)d_in[1];
    const float* tok_emb = (const float*)d_in[2];
    const float* pos_emb = (const float*)d_in[3];
    const float* Wq  = (const float*)d_in[4];
    const float* bq  = (const float*)d_in[5];
    const float* Wk  = (const float*)d_in[6];
    const float* bk  = (const float*)d_in[7];
    const float* Wv  = (const float*)d_in[8];
    const float* bv  = (const float*)d_in[9];
    const float* lng = (const float*)d_in[10];
    const float* lnb = (const float*)d_in[11];
    const float* W1  = (const float*)d_in[12];
    const float* b1  = (const float*)d_in[13];
    const float* W2  = (const float*)d_in[14];
    const float* b2  = (const float*)d_in[15];
    const float* f_g = (const float*)d_in[16];
    const float* f_b = (const float*)d_in[17];
    const float* Wcls = (const float*)d_in[18];
    const float* bcls = (const float*)d_in[19];

    // ---- workspace layout (units: floats; bf16 regions hold 2 elems/float) ----
    float* ws = (float*)d_ws;
    const long sz_h  = M * D;          // 6,291,456
    const long szb   = sz_h / 2;       // bf16 [M,D] in float units
    long off = 0;
    float*  h    = ws + off;            off += sz_h;
    __bf16* hb   = (__bf16*)(ws + off); off += szb;
    __bf16* qb   = (__bf16*)(ws + off); off += szb;
    __bf16* kb   = (__bf16*)(ws + off); off += szb;
    __bf16* vb   = (__bf16*)(ws + off); off += szb;
    __bf16* vT   = (__bf16*)(ws + off); off += szb;   // [B*H, DH, S] == M*D elems
    float*  sc   = ws + off;            off += (long)S * S;
    __bf16* scb  = (__bf16*)(ws + off); off += (long)S * S / 2;
    __bf16* f1b  = (__bf16*)(ws + off); off += M * FF / 2;
    float*  aout = ws + off;            off += sz_h;
    __bf16* wT   = (__bf16*)(ws + off); off += (long)FF * D / 2;  // max weight^T
    float*  rowM = ws + off;            off += M;
    float*  rowS = ws + off;            off += M;
    float*  rowL = ws + off;            off += M;
    // aliases (time-disjoint): classifier chunk over qb..vT, Wcls^T over sc..aout
    float*  chunk = (float*)qb;                       // 8192*1280 fp32 fits
    __bf16* wclsT = (__bf16*)sc;                      // 32000*768 bf16 fits
    const int CN = 1280;                              // 32000 = 25*1280, 1280%64==0
    if (ws_size < (size_t)off * sizeof(float)) return;

    dim3 blk(256);

    auto gemm = [&](const __bf16* A, long lda, long aOff,
                    const __bf16* Bp, long ldb, long bOff,
                    float* Cf, __bf16* Cb, long ldc, long cOff,
                    int Mg, int Ng, int Kg, const float* bias,
                    float scale, int relu, int maskMode, const int* tok) {
        dim3 grid(Mg / BM, Ng / BN);
        gemm_bf16_wmma<<<grid, blk, 0, stream>>>(A, lda, aOff, Bp, ldb, bOff,
                                                 Cf, Cb, ldc, cOff, Kg, bias,
                                                 scale, relu, maskMode, tok);
    };
    auto transW = [&](const float* src, int R, int C, __bf16* dst) {
        transpose_f32_bf16<<<dim3(C / 32, R / 32), blk, 0, stream>>>(src, C, dst, R);
    };

    // embedding
    {
        long total = M * D;
        embed_kernel<<<dim3((unsigned)((total + 255) / 256)), blk, 0, stream>>>(
            x, tok_emb, pos_emb, h, hb, S, D, total);
    }

    for (int i = 0; i < Nl; ++i) {
        for (int j = 0; j < 2; ++j) {
            long wOff = ((long)i * 2 + j) * D * D;
            long vOff = ((long)i * 2 + j) * D;
            // QKV: bf16-only outputs
            transW(Wq + wOff, D, D, wT);
            gemm(hb, D, 0, wT, D, 0, nullptr, qb, D, 0, (int)M, D, D, bq + vOff, 1.f, 0, 0, nullptr);
            transW(Wk + wOff, D, D, wT);
            gemm(hb, D, 0, wT, D, 0, nullptr, kb, D, 0, (int)M, D, D, bk + vOff, 1.f, 0, 0, nullptr);
            transW(Wv + wOff, D, D, wT);
            gemm(hb, D, 0, wT, D, 0, nullptr, vb, D, 0, (int)M, D, D, bv + vOff, 1.f, 0, 0, nullptr);
            // v -> [bh, DH, S] so the A·V GEMM B-operand is [N,K] row-major
            transpose_v_bf16<<<dim3(DH / 32, S / 32, B * H), blk, 0, stream>>>(vb, vT, S, D, DH, H);
            for (int b = 0; b < B; ++b) {
                for (int hh = 0; hh < H; ++hh) {
                    long qkOff = (long)b * S * D + (long)hh * DH;
                    long bh = (long)b * H + hh;
                    // scores = q k^T / 20 with causal|pad mask (fp32 out)
                    gemm(qb, D, qkOff, kb, D, qkOff, sc, nullptr, S, 0,
                         S, S, DH, nullptr, 0.05f, 0, 1, x + (long)b * S);
                    softmax_q_kernel<<<dim3(S / 256), blk, 0, stream>>>(sc, scb, S);
                    // o = a @ v   (A = bf16 softmax, B = vT[bh])
                    gemm(scb, S, 0, vT, S, bh * DH * S, aout, nullptr, D, qkOff,
                         S, DH, S, nullptr, 1.f, 0, 0, nullptr);
                }
            }
            add_ln_kernel<<<dim3((unsigned)M), blk, 0, stream>>>(
                h, aout, lng + vOff, lnb + vOff, h, hb, D);
        }
        // FFN
        transW(W1 + (long)i * D * FF, D, FF, wT);
        gemm(hb, D, 0, wT, D, 0, nullptr, f1b, FF, 0, (int)M, FF, D,
             b1 + (long)i * FF, 1.f, 1, 0, nullptr);
        transW(W2 + (long)i * FF * D, FF, D, wT);
        gemm(f1b, FF, 0, wT, FF, 0, aout, nullptr, D, 0, (int)M, D, FF,
             b2 + (long)i * D, 1.f, 0, 0, nullptr);
        add_ln_kernel<<<dim3((unsigned)M), blk, 0, stream>>>(
            h, aout, f_g + (long)i * D, f_b + (long)i * D, h, hb, D);
    }

    // classifier: Wcls^T once, then chunked GEMM + streaming LSE + loss
    transW(Wcls, D, V, wclsT);
    lse_init_kernel<<<dim3((unsigned)((M + 255) / 256)), blk, 0, stream>>>(rowM, rowS, rowL, (int)M);
    for (int c0 = 0; c0 < V; c0 += CN) {
        gemm(hb, D, 0, wclsT, D, (long)c0 * D, chunk, nullptr, CN, 0,
             (int)M, CN, D, bcls + c0, 1.f, 0, 0, nullptr);
        lse_chunk_kernel<<<dim3((unsigned)M), blk, 0, stream>>>(
            chunk, CN, c0, label, rowM, rowS, rowL);
    }
    loss_kernel<<<1, blk, 0, stream>>>(rowM, rowS, rowL, (float*)d_out, (int)M);
}
